// SlidingWindowCrossAttention_47742856463118
// MI455X (gfx1250) — matrix operand order repaired and checked by reading, not compiled, not run on hardware
//
#include <hip/hip_runtime.h>

#define EMBED  1024
#define HEADS  16
#define HDIM   64
#define SEQ    2048
#define BATCH  2
#define WIN    512

typedef __attribute__((ext_vector_type(16))) __bf16 v16bf;
typedef __attribute__((ext_vector_type(8)))  __bf16 v8bf;
typedef __attribute__((ext_vector_type(8)))  float  v8f;

__device__ __forceinline__ __bf16 f2bf(float f) {
  unsigned u = __builtin_bit_cast(unsigned, f);
  u += 0x7FFFu + ((u >> 16) & 1u);            // round-to-nearest-even
  unsigned short s = (unsigned short)(u >> 16);
  return __builtin_bit_cast(__bf16, s);
}
__device__ __forceinline__ float bf2f(__bf16 b) {
  unsigned short s = __builtin_bit_cast(unsigned short, b);
  unsigned u = ((unsigned)s) << 16;
  return __builtin_bit_cast(float, u);
}
__device__ __forceinline__ v16bf mk16(v8bf lo, v8bf hi) {
  v16bf r;
#pragma unroll
  for (int i = 0; i < 8; ++i) { r[i] = lo[i]; r[i + 8] = hi[i]; }
  return r;
}
__device__ __forceinline__ v8bf shx8(v8bf x) {   // exchange with lane^16
  union { v8bf v; unsigned u[4]; } a; a.v = x;
#pragma unroll
  for (int i = 0; i < 4; ++i) a.u[i] = __shfl_xor(a.u[i], 16, 32);
  return a.v;
}

// ---------------------------------------------------------------- f32 -> bf16
__global__ void __launch_bounds__(256)
cvt_kernel(const float* __restrict__ s, __bf16* __restrict__ d, int n4) {
  int i = blockIdx.x * 256 + threadIdx.x;
  if (i >= n4) return;
  const float4 f = ((const float4*)s)[i];
  union { __bf16 h[4]; unsigned long long u; } o;
  o.h[0] = f2bf(f.x); o.h[1] = f2bf(f.y); o.h[2] = f2bf(f.z); o.h[3] = f2bf(f.w);
  ((unsigned long long*)d)[i] = o.u;
}

// ------------------------------------------------- Y = X * W^T + bias (WMMA)
// X: [M,K] bf16 row-major, W: [N,K] bf16 row-major.
// Wave tile: 64(M) x 32(N) -> 8 WMMA per 32-K step at 1.5 b128 loads/WMMA.
// Block = 4 waves -> 64 x 128 tile.
__global__ void __launch_bounds__(128)
gemm_bf16_kernel(const __bf16* __restrict__ X, const __bf16* __restrict__ Wm,
                 const float* __restrict__ bias, __bf16* __restrict__ Yb,
                 float* __restrict__ Yf, int K, int N)
{
  const int lane = threadIdx.x & 31, wave = threadIdx.x >> 5;
  const int grp = lane >> 4, li = lane & 15;
  const int m0 = blockIdx.x * 64;
  const int n0 = blockIdx.y * 128 + wave * 32;
  const __bf16* pa  = X  + (size_t)(m0 + li) * K + grp * 16;  // A frag chunks
  const __bf16* pb0 = Wm + (size_t)(n0 + li) * K + grp * 16;  // B frag: 16 contig K
  const __bf16* pb1 = pb0 + (size_t)16 * K;
  v8f c[4][2] = {};
  for (int k0 = 0; k0 < K; k0 += 32) {
    __builtin_prefetch((const void*)(pb0 + k0 + 256), 0, 3);
    __builtin_prefetch((const void*)(pa  + k0 + 256), 0, 3);
    v16bf b0 = *(const v16bf*)(pb0 + k0);
    v16bf b1 = *(const v16bf*)(pb1 + k0);
#pragma unroll
    for (int i = 0; i < 4; ++i) {
      const __bf16* pai = pa + (size_t)(i * 16) * K + k0;
      v16bf a = mk16(*(const v8bf*)(pai), *(const v8bf*)(pai + 8));
      c[i][0] = __builtin_amdgcn_wmma_f32_16x16x32_bf16(false, a, false, b0, (short)0, c[i][0], false, false);
      c[i][1] = __builtin_amdgcn_wmma_f32_16x16x32_bf16(false, a, false, b1, (short)0, c[i][1], false, false);
    }
  }
#pragma unroll
  for (int j = 0; j < 2; ++j) {
    const int col = n0 + j * 16 + li;
    const float bn = bias[col];
#pragma unroll
    for (int i = 0; i < 4; ++i) {
      const int r0 = m0 + i * 16 + 8 * grp;
      if (Yf) {
#pragma unroll
        for (int v = 0; v < 8; ++v)
          Yf[(size_t)(r0 + v) * N + col] = c[i][j][v] + bn;
      } else {
#pragma unroll
        for (int v = 0; v < 8; ++v)
          Yb[(size_t)(r0 + v) * N + col] = f2bf(c[i][j][v] + bn);
      }
    }
  }
}

// ------------------------------------------------------------- RoPE in-place
// X layout [b, s, h*64 + d]; rotates (d, d+32) pairs per head, then scales.
__global__ void __launch_bounds__(256)
rope_kernel(__bf16* __restrict__ X, float scale) {
  unsigned tid = blockIdx.x * 256 + threadIdx.x;   // BATCH*SEQ*HEADS*32 threads
  int d = tid & 31;
  unsigned r = tid >> 5;
  int h = r & (HEADS - 1);
  unsigned bs = r >> 4;                             // b*SEQ + s
  int s = bs & (SEQ - 1);
  size_t idx = (size_t)bs * EMBED + (size_t)h * HDIM + d;
  float x1 = bf2f(X[idx]), x2 = bf2f(X[idx + 32]);
  float inv = __powf(10000.0f, -(float)d * (1.0f / 32.0f));
  float ang = (float)s * inv;
  float sn, cs;
  __sincosf(ang, &sn, &cs);
  X[idx]      = f2bf((x1 * cs - x2 * sn) * scale);
  X[idx + 32] = f2bf((x1 * sn + x2 * cs) * scale);
}

// -------------------------------------------- sliding-window flash attention
// One wave handles a 16-query tile of one (b,h). Computes S^T = K*Q^T so each
// lane owns one query column; streaming softmax; out^T = V^T * P^T with V^T
// staged through a per-wave LDS transpose. Output written head-merged bf16.
__global__ void __launch_bounds__(256)
attn_kernel(const __bf16* __restrict__ Qm, const __bf16* __restrict__ Km,
            const __bf16* __restrict__ Vm, __bf16* __restrict__ Om)
{
  __shared__ __attribute__((aligned(16))) __bf16 lds[8 * 64 * 40];  // 40 KB
  const int lane = threadIdx.x & 31, wave = threadIdx.x >> 5;
  const int grp = lane >> 4, li = lane & 15;
  const int gw = blockIdx.x * 8 + wave;       // global wave id
  const int qt = gw & (SEQ / 16 - 1);         // 128 q-tiles
  const int h  = (gw >> 7) & (HEADS - 1);
  const int b  = gw >> 11;
  const int q0 = qt * 16;
  const int q  = q0 + li;                     // this lane's query index
  const __bf16* Qb = Qm + ((size_t)b * SEQ + q0) * EMBED + (size_t)h * HDIM;
  const __bf16* Kb = Km + (size_t)b * SEQ * EMBED + (size_t)h * HDIM;
  const __bf16* Vb = Vm + (size_t)b * SEQ * EMBED + (size_t)h * HDIM;
  __bf16* vt = lds + wave * 64 * 40;

  // B-fragments of Q^T (kd x q): lane holds 16 contiguous head-dim values.
  const v16bf bq0 = *(const v16bf*)(Qb + (size_t)li * EMBED + grp * 16);
  const v16bf bq1 = *(const v16bf*)(Qb + (size_t)li * EMBED + 32 + grp * 16);

  float m = -1e30f, l = 0.0f;
  v8f acc[4] = {{}, {}, {}, {}};

  int ks = q0 - WIN; if (ks < 0) ks = 0; ks &= ~31;
  for (int kt = ks; kt <= q0 + 15; kt += 32) {
    // ---- S^T: two 16-key tiles, each 2 WMMA over head dim 64
    int kr0 = kt + li;        if (kr0 > SEQ - 1) kr0 = SEQ - 1;
    int kr1 = kt + 16 + li;   if (kr1 > SEQ - 1) kr1 = SEQ - 1;
    const __bf16* kp0 = Kb + (size_t)kr0 * EMBED;
    const __bf16* kp1 = Kb + (size_t)kr1 * EMBED;
    v8f z = {};
    v16bf ak;
    ak = mk16(*(const v8bf*)(kp0 + grp * 16), *(const v8bf*)(kp0 + 8 + grp * 16));
    v8f s0 = __builtin_amdgcn_wmma_f32_16x16x32_bf16(false, ak, false, bq0, (short)0, z, false, false);
    ak = mk16(*(const v8bf*)(kp0 + 32 + grp * 16), *(const v8bf*)(kp0 + 40 + grp * 16));
    s0 = __builtin_amdgcn_wmma_f32_16x16x32_bf16(false, ak, false, bq1, (short)0, s0, false, false);
    ak = mk16(*(const v8bf*)(kp1 + grp * 16), *(const v8bf*)(kp1 + 8 + grp * 16));
    v8f s1 = __builtin_amdgcn_wmma_f32_16x16x32_bf16(false, ak, false, bq0, (short)0, z, false, false);
    ak = mk16(*(const v8bf*)(kp1 + 32 + grp * 16), *(const v8bf*)(kp1 + 40 + grp * 16));
    s1 = __builtin_amdgcn_wmma_f32_16x16x32_bf16(false, ak, false, bq1, (short)0, s1, false, false);

    // ---- stage V tile (32 keys x 64 dims) transposed into LDS
    int kv = kt + lane; if (kv > SEQ - 1) kv = SEQ - 1;
    const __bf16* vp = Vb + (size_t)kv * EMBED;
#pragma unroll
    for (int c = 0; c < 8; ++c) {
      v8bf vv = *(const v8bf*)(vp + c * 8);
#pragma unroll
      for (int j = 0; j < 8; ++j)
        vt[(c * 8 + j) * 40 + lane] = vv[j];
    }

    // ---- mask + streaming softmax (per-lane query, shfl over lane^16)
    float mx = -1e30f;
#pragma unroll
    for (int v = 0; v < 8; ++v) {
      int k0i = kt + 8 * grp + v;
      int k1i = kt + 16 + 8 * grp + v;
      float a0 = s0[v]; if (k0i > q || k0i < q - WIN) a0 = -1e30f;
      float a1 = s1[v]; if (k1i > q || k1i < q - WIN) a1 = -1e30f;
      s0[v] = a0; s1[v] = a1;
      mx = fmaxf(mx, fmaxf(a0, a1));
    }
    mx = fmaxf(mx, __shfl_xor(mx, 16, 32));
    float mn = fmaxf(m, mx);
    float alpha = __expf(m - mn);
    float rs = 0.0f;
    v8bf pb0, pb1;
#pragma unroll
    for (int v = 0; v < 8; ++v) {
      float e0 = (s0[v] < -1e29f) ? 0.0f : __expf(s0[v] - mn);
      float e1 = (s1[v] < -1e29f) ? 0.0f : __expf(s1[v] - mn);
      rs += e0 + e1;
      pb0[v] = f2bf(e0); pb1[v] = f2bf(e1);
    }
    rs += __shfl_xor(rs, 16, 32);
    l = l * alpha + rs;
    m = mn;
#pragma unroll
    for (int f = 0; f < 4; ++f) acc[f] = acc[f] * alpha;

    // ---- build P^T B-fragment (k x q) from score fragments
    v8bf sh0 = shx8(pb0), sh1 = shx8(pb1);
    v16bf pf = grp ? mk16(sh1, pb1) : mk16(pb0, sh0);

    // ---- out^T += V^T * P^T  (4 fragments cover head dim 64)
#pragma unroll
    for (int f = 0; f < 4; ++f) {
      const __bf16* vd = vt + (f * 16 + li) * 40;
      v16bf av = mk16(*(const v8bf*)(vd + grp * 16), *(const v8bf*)(vd + 8 + grp * 16));
      acc[f] = __builtin_amdgcn_wmma_f32_16x16x32_bf16(false, av, false, pf, (short)0, acc[f], false, false);
    }
  }

  // ---- normalize and store head-merged bf16 [b, q, h*64 + d]
  float rinv = (l > 0.0f) ? 1.0f / l : 0.0f;
  __bf16* ob = Om + ((size_t)b * SEQ + q) * EMBED + (size_t)h * HDIM + grp * 8;
#pragma unroll
  for (int f = 0; f < 4; ++f) {
    v8bf o;
#pragma unroll
    for (int v = 0; v < 8; ++v) o[v] = f2bf(acc[f][v] * rinv);
    *(v8bf*)(ob + f * 16) = o;
  }
}

// ---------------------------------------------------------------------- host
extern "C" void kernel_launch(void* const* d_in, const int* in_sizes, int n_in,
                              void* d_out, int out_size, void* d_ws, size_t ws_size,
                              hipStream_t stream)
{
  (void)in_sizes; (void)n_in; (void)out_size; (void)ws_size;
  const float* query = (const float*)d_in[0];
  const float* keyi  = (const float*)d_in[1];
  const float* vali  = (const float*)d_in[2];
  const float* Wq = (const float*)d_in[3];  const float* bq = (const float*)d_in[4];
  const float* Wk = (const float*)d_in[5];  const float* bk = (const float*)d_in[6];
  const float* Wv = (const float*)d_in[7];  const float* bv = (const float*)d_in[8];
  const float* Wo = (const float*)d_in[9];  const float* bo = (const float*)d_in[10];

  const size_t TOK = (size_t)BATCH * SEQ;       // 4096 rows
  const size_t TEN = TOK * EMBED;               // activation elements
  const size_t WEI = (size_t)EMBED * EMBED;     // weight elements
  char* ws = (char*)d_ws;
  __bf16* xq  = (__bf16*)ws; ws += TEN * 2;
  __bf16* xk  = (__bf16*)ws; ws += TEN * 2;
  __bf16* xv  = (__bf16*)ws; ws += TEN * 2;
  __bf16* wqb = (__bf16*)ws; ws += WEI * 2;
  __bf16* wkb = (__bf16*)ws; ws += WEI * 2;
  __bf16* wvb = (__bf16*)ws; ws += WEI * 2;
  __bf16* wob = (__bf16*)ws; ws += WEI * 2;
  __bf16* qp  = (__bf16*)ws; ws += TEN * 2;
  __bf16* kp  = (__bf16*)ws; ws += TEN * 2;
  __bf16* vp  = (__bf16*)ws; ws += TEN * 2;
  __bf16* ab  = (__bf16*)ws; ws += TEN * 2;

  const int tn4 = (int)(TEN / 4), wn4 = (int)(WEI / 4);
  cvt_kernel<<<tn4 / 256, 256, 0, stream>>>(query, xq, tn4);
  cvt_kernel<<<tn4 / 256, 256, 0, stream>>>(keyi,  xk, tn4);
  cvt_kernel<<<tn4 / 256, 256, 0, stream>>>(vali,  xv, tn4);
  cvt_kernel<<<wn4 / 256, 256, 0, stream>>>(Wq, wqb, wn4);
  cvt_kernel<<<wn4 / 256, 256, 0, stream>>>(Wk, wkb, wn4);
  cvt_kernel<<<wn4 / 256, 256, 0, stream>>>(Wv, wvb, wn4);
  cvt_kernel<<<wn4 / 256, 256, 0, stream>>>(Wo, wob, wn4);

  dim3 ggrid((unsigned)(TOK / 64), EMBED / 128);
  gemm_bf16_kernel<<<ggrid, 128, 0, stream>>>(xq, wqb, bq, qp, nullptr, EMBED, EMBED);
  gemm_bf16_kernel<<<ggrid, 128, 0, stream>>>(xk, wkb, bk, kp, nullptr, EMBED, EMBED);
  gemm_bf16_kernel<<<ggrid, 128, 0, stream>>>(xv, wvb, bv, vp, nullptr, EMBED, EMBED);

  const int rth = BATCH * SEQ * HEADS * 32;     // rope threads
  rope_kernel<<<rth / 256, 256, 0, stream>>>(qp, 0.125f);  // fold Dh^-0.5
  rope_kernel<<<rth / 256, 256, 0, stream>>>(kp, 1.0f);

  const int waves = BATCH * HEADS * (SEQ / 16);
  attn_kernel<<<waves / 8, 256, 0, stream>>>(qp, kp, vp, ab);

  gemm_bf16_kernel<<<ggrid, 128, 0, stream>>>(ab, wob, bo, nullptr, (float*)d_out, EMBED, EMBED);
}
